// TransformerDecoderCell_17970143166769
// MI455X (gfx1250) — compile-verified, matrix operand order checked by reading, not run on hardware
//
#include <hip/hip_runtime.h>

typedef __attribute__((ext_vector_type(16))) _Float16 v16h;
typedef __attribute__((ext_vector_type(8)))  _Float16 v8h;
typedef __attribute__((ext_vector_type(8)))  float    v8f;
typedef _Float16 half_t;

static __device__ inline v8f wmma32(v16h a, v16h b, v8f c) {
  // D = A(16x32 f16) * B(32x16 f16) + C(16x16 f32)
  return __builtin_amdgcn_wmma_f32_16x16x32_f16(false, a, false, b, (short)0, c, false, false);
}

// Load a 16x32 f16 fragment (A/B operand layout) from a row-major tile.
// base points at (row0, col0); ld = row stride in elements.
// Lane l holds row (l&15); K elems: [(l>>4)*8 .. +8) and [16+(l>>4)*8 .. +8).
static __device__ inline v16h load_tile_a(const half_t* __restrict__ base, int ld,
                                          int kbase, int lane) {
  const int r = lane & 15;
  const int s = (lane >> 4) << 3;  // 0 or 8
  const half_t* p = base + (size_t)r * ld + kbase + s;
  v16h out;
  *(v8h*)&out       = *(const v8h*)p;
  ((v8h*)&out)[1]   = *(const v8h*)(p + 16);
  return out;
}

// ---------------------------------------------------------------------------
// f32 -> f16 conversion
// ---------------------------------------------------------------------------
__global__ __launch_bounds__(256)
void cvt_f32_f16_kernel(const float* __restrict__ in, half_t* __restrict__ out, int n) {
  int i = blockIdx.x * 256 + threadIdx.x;
  if (i < n) out[i] = (half_t)in[i];
}

// ---------------------------------------------------------------------------
// GEMM: out = epilogue( A[M,K] @ Bw[N,K]^T + bias )   (x @ W.T convention)
// Block: 256 thr = 8 waves (2x4), wave tile 32x32 (2x2 WMMA), block tile 64x128.
// epilogue: t=(dot+bias)*scale; relu?; +resid?; store f32 and/or f16.
// M,N,K are multiples of 64/128/32 here; no bounds checks needed.
// ---------------------------------------------------------------------------
__global__ __launch_bounds__(256)
void gemm_bias_kernel(const half_t* __restrict__ A, const half_t* __restrict__ Bw,
                      const float* __restrict__ bias, const float* __restrict__ resid,
                      float* __restrict__ outF, half_t* __restrict__ outH,
                      int M, int N, int K, float scale, int relu) {
  const int lane = threadIdx.x & 31;
  const int w    = threadIdx.x >> 5;
  const int wm   = w >> 2, wn = w & 3;
  const int row0 = blockIdx.y * 64 + wm * 32;
  const int col0 = blockIdx.x * 128 + wn * 32;

  v8f c00 = {}, c01 = {}, c10 = {}, c11 = {};
  const half_t* Ar0 = A  + (size_t)row0 * K;
  const half_t* Ar1 = A  + (size_t)(row0 + 16) * K;
  const half_t* Br0 = Bw + (size_t)col0 * K;
  const half_t* Br1 = Bw + (size_t)(col0 + 16) * K;

  for (int kb = 0; kb < K; kb += 32) {
    v16h a0 = load_tile_a(Ar0, K, kb, lane);
    v16h a1 = load_tile_a(Ar1, K, kb, lane);
    v16h b0 = load_tile_a(Br0, K, kb, lane);
    v16h b1 = load_tile_a(Br1, K, kb, lane);
    c00 = wmma32(a0, b0, c00);
    c01 = wmma32(a0, b1, c01);
    c10 = wmma32(a1, b0, c10);
    c11 = wmma32(a1, b1, c11);
  }

  const int n  = lane & 15;
  const int hi = lane >> 4;
  const float bn0 = bias ? bias[col0 + n]      : 0.f;
  const float bn1 = bias ? bias[col0 + 16 + n] : 0.f;

  for (int r = 0; r < 8; ++r) {
    const int rr0 = row0 + r + hi * 8;
    const int rr1 = rr0 + 16;
    float vals[4] = { c00[r] + bn0, c01[r] + bn1, c10[r] + bn0, c11[r] + bn1 };
    int   rows[4] = { rr0, rr0, rr1, rr1 };
    int   cols[4] = { col0 + n, col0 + 16 + n, col0 + n, col0 + 16 + n };
    for (int t = 0; t < 4; ++t) {
      float x = vals[t] * scale;
      if (relu) x = fmaxf(x, 0.f);
      size_t idx = (size_t)rows[t] * N + cols[t];
      if (resid) x += resid[idx];
      if (outF) outF[idx] = x;
      if (outH) outH[idx] = (half_t)x;
    }
  }
}

// ---------------------------------------------------------------------------
// Flash-style attention. Q pre-scaled by 1/sqrt(D). Layout [B, L, H*dh].
// Block = 128 thr = 4 waves; each wave owns one 16-row Q tile (block: 64 rows).
// Pass 1: online row max / exp-sum (shuffle reductions in 16-lane halves).
// Pass 2: recompute S, P=exp(S-m)/l -> LDS transpose -> WMMA P @ V.
// ---------------------------------------------------------------------------
__global__ __launch_bounds__(128)
void attn_kernel(const half_t* __restrict__ Q, const half_t* __restrict__ Km,
                 const half_t* __restrict__ V, half_t* __restrict__ O, int causal) {
  const int L = 1024, D = 1024, DH = 64, HEADS = 16;
  __shared__ __align__(16) half_t ldsV[32 * 64];        // 32 keys x 64 dh
  __shared__ __align__(16) half_t ldsP[4][16 * 32];     // per-wave P scratch

  const int lane = threadIdx.x & 31;
  const int wv   = threadIdx.x >> 5;
  const int nqb  = L / 64;
  const int bh   = blockIdx.x / nqb;
  const int qb   = blockIdx.x % nqb;
  const int b    = bh / HEADS, h = bh % HEADS;
  const int q0   = qb * 64 + wv * 16;
  const int n    = lane & 15;
  const int hi   = lane >> 4;

  const half_t* Qb = Q  + (size_t)(b * L) * D + h * DH;
  const half_t* Kb = Km + (size_t)(b * L) * D + h * DH;
  const half_t* Vb = V  + (size_t)(b * L) * D + h * DH;

  v16h qa0 = load_tile_a(Qb + (size_t)q0 * D, D, 0, lane);
  v16h qa1 = load_tile_a(Qb + (size_t)q0 * D, D, 32, lane);

  // ---- Pass 1: row max + exp-sum ----
  float rowm[8], rsum[8];
  for (int r = 0; r < 8; ++r) { rowm[r] = -1e30f; rsum[r] = 0.f; }

  for (int kk = 0; kk < L; kk += 16) {
    if (causal && kk > q0 + 15) break;
    v16h kb0 = load_tile_a(Kb + (size_t)kk * D, D, 0, lane);
    v16h kb1 = load_tile_a(Kb + (size_t)kk * D, D, 32, lane);
    v8f s = {};
    s = wmma32(qa0, kb0, s);
    s = wmma32(qa1, kb1, s);
    const int coln = kk + n;
    for (int r = 0; r < 8; ++r) {
      const int row = q0 + r + hi * 8;
      float sv = s[r];
      if (causal && coln > row) sv = -1e30f;
      float mx = sv;
      mx = fmaxf(mx, __shfl_xor(mx, 1));
      mx = fmaxf(mx, __shfl_xor(mx, 2));
      mx = fmaxf(mx, __shfl_xor(mx, 4));
      mx = fmaxf(mx, __shfl_xor(mx, 8));
      const float mnew = fmaxf(rowm[r], mx);
      float e = __expf(sv - mnew);
      e += __shfl_xor(e, 1);
      e += __shfl_xor(e, 2);
      e += __shfl_xor(e, 4);
      e += __shfl_xor(e, 8);
      rsum[r] = rsum[r] * __expf(rowm[r] - mnew) + e;
      rowm[r] = mnew;
    }
  }
  float rinv[8];
  for (int r = 0; r < 8; ++r) rinv[r] = 1.f / rsum[r];

  // ---- Pass 2: P @ V ----
  v8f o0 = {}, o1 = {}, o2 = {}, o3 = {};
  const int kend = causal ? (qb * 64 + 64) : L;   // uniform over the block

  for (int kk = 0; kk < kend; kk += 32) {
    __syncthreads();  // protect ldsV from previous iteration's readers
    {   // stage V[kk..kk+31][0..63] cooperatively (128 threads x 16 halves)
      const int t  = threadIdx.x;
      const int vr = t >> 2;
      const int vc = (t & 3) * 16;
      const half_t* src = Vb + (size_t)(kk + vr) * D + vc;
      *(v8h*)&ldsV[vr * 64 + vc]     = *(const v8h*)src;
      *(v8h*)&ldsV[vr * 64 + vc + 8] = *(const v8h*)(src + 8);
    }
    __syncthreads();

    // compute P for two 16-key tiles, write to per-wave LDS scratch
    for (int t2 = 0; t2 < 2; ++t2) {
      const int kt = kk + t2 * 16;
      v16h kb0 = load_tile_a(Kb + (size_t)kt * D, D, 0, lane);
      v16h kb1 = load_tile_a(Kb + (size_t)kt * D, D, 32, lane);
      v8f s = {};
      s = wmma32(qa0, kb0, s);
      s = wmma32(qa1, kb1, s);
      const int coln = kt + n;
      for (int r = 0; r < 8; ++r) {
        const int row = q0 + r + hi * 8;
        float sv = s[r];
        if (causal && coln > row) sv = -1e30f;
        const float p = __expf(sv - rowm[r]) * rinv[r];
        ldsP[wv][(r + hi * 8) * 32 + t2 * 16 + n] = (half_t)p;
      }
    }
    __syncthreads();

    // gather P (A layout) from LDS
    v16h pa;
    {
      const half_t* pp = &ldsP[wv][0] + (size_t)(lane & 15) * 32 + (hi << 3);
      *(v8h*)&pa     = *(const v8h*)pp;
      ((v8h*)&pa)[1] = *(const v8h*)(pp + 16);
    }
    // gather V columns (B layout) and accumulate O
    const int b0 = hi * 8;
    for (int c = 0; c < 4; ++c) {
      v16h vbt;
      for (int e = 0; e < 16; ++e) {
        const int kidx = (e < 8) ? (b0 + e) : (16 + b0 + (e - 8));
        vbt[e] = ldsV[kidx * 64 + c * 16 + n];
      }
      if      (c == 0) o0 = wmma32(pa, vbt, o0);
      else if (c == 1) o1 = wmma32(pa, vbt, o1);
      else if (c == 2) o2 = wmma32(pa, vbt, o2);
      else             o3 = wmma32(pa, vbt, o3);
    }
  }

  half_t* Ob = O + (size_t)(b * L) * D + h * DH;
  for (int r = 0; r < 8; ++r) {
    const int row = q0 + r + hi * 8;
    Ob[(size_t)row * D +  0 + n] = (half_t)o0[r];
    Ob[(size_t)row * D + 16 + n] = (half_t)o1[r];
    Ob[(size_t)row * D + 32 + n] = (half_t)o2[r];
    Ob[(size_t)row * D + 48 + n] = (half_t)o3[r];
  }
}

// ---------------------------------------------------------------------------
// LayerNorm over last dim (D=1024), one row per block. Optional f16 mirror.
// ---------------------------------------------------------------------------
__global__ __launch_bounds__(256)
void ln_kernel(const float* __restrict__ x, const float* __restrict__ g,
               const float* __restrict__ b, float* __restrict__ outF,
               half_t* __restrict__ outH, int D) {
  __shared__ float red[256];
  const int row = blockIdx.x;
  const float* xr = x + (size_t)row * D;

  float s = 0.f, ss = 0.f;
  for (int i = threadIdx.x; i < D; i += 256) {
    float v = xr[i];
    s += v; ss += v * v;
  }
  red[threadIdx.x] = s; __syncthreads();
  for (int off = 128; off > 0; off >>= 1) {
    if (threadIdx.x < off) red[threadIdx.x] += red[threadIdx.x + off];
    __syncthreads();
  }
  const float mean = red[0] / D;
  __syncthreads();
  red[threadIdx.x] = ss; __syncthreads();
  for (int off = 128; off > 0; off >>= 1) {
    if (threadIdx.x < off) red[threadIdx.x] += red[threadIdx.x + off];
    __syncthreads();
  }
  const float var = red[0] / D - mean * mean;
  const float rs  = rsqrtf(var + 1e-5f);

  for (int i = threadIdx.x; i < D; i += 256) {
    float y = (xr[i] - mean) * rs * g[i] + b[i];
    outF[(size_t)row * D + i] = y;
    if (outH) outH[(size_t)row * D + i] = (half_t)y;
  }
}

// ---------------------------------------------------------------------------
extern "C" void kernel_launch(void* const* d_in, const int* in_sizes, int n_in,
                              void* d_out, int out_size, void* d_ws, size_t ws_size,
                              hipStream_t stream) {
  (void)in_sizes; (void)n_in; (void)out_size; (void)ws_size;
  const int B = 4, L = 1024, Dm = 1024, FF = 4096;
  const int M = B * L;                 // 4096 rows
  const float invSqrtD = 1.0f / 32.0f; // 1/sqrt(1024)

  const float* x   = (const float*)d_in[0];
  const float* enc = (const float*)d_in[1];
  // d_in[2]=src_mask (all ones), d_in[3]=tgt_mask (tril): encoded via causal flag
  const float* wq = (const float*)d_in[4];  const float* bq = (const float*)d_in[5];
  const float* wk = (const float*)d_in[6];  const float* bk = (const float*)d_in[7];
  const float* wv = (const float*)d_in[8];  const float* bv = (const float*)d_in[9];
  const float* wo = (const float*)d_in[10]; const float* bo = (const float*)d_in[11];
  const float* w1 = (const float*)d_in[12]; const float* b1 = (const float*)d_in[13];
  const float* w2 = (const float*)d_in[14]; const float* b2 = (const float*)d_in[15];
  const float* g1 = (const float*)d_in[16]; const float* be1 = (const float*)d_in[17];
  const float* g2 = (const float*)d_in[18]; const float* be2 = (const float*)d_in[19];
  float* out = (float*)d_out;

  const size_t BLD = (size_t)M * Dm;   // 4 Mi elems
  const size_t DD  = (size_t)Dm * Dm;  // 1 Mi
  const size_t FD  = (size_t)FF * Dm;  // 4 Mi
  const size_t BLF = (size_t)M * FF;   // 16 Mi

  char* ws = (char*)d_ws;
  size_t off = 0;
  auto alloc = [&](size_t bytes) -> char* {
    char* p = ws + off;
    off += (bytes + 255) & ~(size_t)255;
    return p;
  };
  half_t* xh   = (half_t*)alloc(BLD * 2);
  half_t* ench = (half_t*)alloc(BLD * 2);
  half_t* wqh  = (half_t*)alloc(DD * 2);
  half_t* wkh  = (half_t*)alloc(DD * 2);
  half_t* wvh  = (half_t*)alloc(DD * 2);
  half_t* woh  = (half_t*)alloc(DD * 2);
  half_t* w1h  = (half_t*)alloc(FD * 2);
  half_t* w2h  = (half_t*)alloc(FD * 2);
  half_t* qh   = (half_t*)alloc(BLD * 2);
  half_t* kh   = (half_t*)alloc(BLD * 2);
  half_t* vh   = (half_t*)alloc(BLD * 2);
  half_t* ah   = (half_t*)alloc(BLD * 2);
  half_t* y1h  = (half_t*)alloc(BLD * 2);
  half_t* y2h  = (half_t*)alloc(BLD * 2);
  half_t* ffh  = (half_t*)alloc(BLF * 2);
  float* y1pre = (float*)alloc(BLD * 4);
  float* y1f   = (float*)alloc(BLD * 4);
  float* y2pre = (float*)alloc(BLD * 4);
  float* y2f   = (float*)alloc(BLD * 4);
  float* ypre  = (float*)alloc(BLD * 4);

  auto cvt = [&](const float* src, half_t* dst, size_t nelem) {
    cvt_f32_f16_kernel<<<dim3((unsigned)((nelem + 255) / 256)), dim3(256), 0, stream>>>(
        src, dst, (int)nelem);
  };
  auto gemm = [&](const half_t* A, const half_t* Bw, const float* bias,
                  const float* resid, float* oF, half_t* oH,
                  int Mm, int Nn, int Kk, float scale, int relu) {
    gemm_bias_kernel<<<dim3(Nn / 128, Mm / 64), dim3(256), 0, stream>>>(
        A, Bw, bias, resid, oF, oH, Mm, Nn, Kk, scale, relu);
  };
  auto attn = [&](const half_t* Qp, const half_t* Kp, const half_t* Vp,
                  half_t* Op, int causal) {
    attn_kernel<<<dim3(B * 16 * (L / 64)), dim3(128), 0, stream>>>(Qp, Kp, Vp, Op, causal);
  };
  auto lnorm = [&](const float* src, const float* gg, const float* bb,
                   float* oF, half_t* oH) {
    ln_kernel<<<dim3(M), dim3(256), 0, stream>>>(src, gg, bb, oF, oH, Dm);
  };

  // f32 -> f16 staging (weights stay resident in the 192MB L2 afterwards)
  cvt(x, xh, BLD);     cvt(enc, ench, BLD);
  cvt(wq, wqh, DD);    cvt(wk, wkh, DD);
  cvt(wv, wvh, DD);    cvt(wo, woh, DD);
  cvt(w1, w1h, FD);    cvt(w2, w2h, FD);

  // ---- self attention (causal) ----
  gemm(xh, wqh, bq, nullptr, nullptr, qh, M, Dm, Dm, invSqrtD, 0); // Q pre-scaled
  gemm(xh, wkh, bk, nullptr, nullptr, kh, M, Dm, Dm, 1.f, 0);
  gemm(xh, wvh, bv, nullptr, nullptr, vh, M, Dm, Dm, 1.f, 0);
  attn(qh, kh, vh, ah, 1);
  gemm(ah, woh, bo, x, y1pre, nullptr, M, Dm, Dm, 1.f, 0);        // + residual x
  lnorm(y1pre, g1, be1, y1f, y1h);                                 // y1 (f32 + f16)

  // ---- cross attention (reuses attn1 weights, per reference) ----
  gemm(y1h,  wqh, bq, nullptr, nullptr, qh, M, Dm, Dm, invSqrtD, 0);
  gemm(ench, wkh, bk, nullptr, nullptr, kh, M, Dm, Dm, 1.f, 0);
  gemm(ench, wvh, bv, nullptr, nullptr, vh, M, Dm, Dm, 1.f, 0);
  attn(qh, kh, vh, ah, 0);
  gemm(ah, woh, bo, y1f, y2pre, nullptr, M, Dm, Dm, 1.f, 0);      // + residual y1
  lnorm(y2pre, g1, be1, y2f, y2h);                                 // y2 (f32 + f16)

  // ---- FFN ----
  gemm(y2h, w1h, b1, nullptr, nullptr, ffh, M, FF, Dm, 1.f, 1);   // ReLU
  gemm(ffh, w2h, b2, y2f, ypre, nullptr, M, Dm, FF, 1.f, 0);      // + residual y2
  lnorm(ypre, g2, be2, out, nullptr);                              // final -> d_out
}
